// FretboardTransformer_69226282877575
// MI455X (gfx1250) — matrix operand order repaired and checked by reading, not compiled, not run on hardware
//
#include <hip/hip_runtime.h>
#include <hip/hip_bf16.h>

// ---------------------------------------------------------------------------
// CDNA5 (gfx1250) FretboardTransformer forward pass.
// f32 residual stream; bf16 WMMA (v_wmma_f32_16x16x32_bf16) for all GEMMs and
// attention. Weights transpose-converted f32(KxN) -> bf16(NxK) once per launch.
// GEMM: 128x128 block, 64x32 wave tile, BK=64, double-buffered LDS staged via
// CDNA5 async global->LDS copies (ASYNCcnt) when available, padded stride 72
// (conflict-free fragment reads), clamped (branchless) ragged-N handling.
// Flash attention: online-softmax, WMMA for both matmuls.
// ---------------------------------------------------------------------------

typedef __bf16 bf16;
typedef __attribute__((ext_vector_type(8)))  __bf16 v8bf;
typedef __attribute__((ext_vector_type(16))) __bf16 v16bf;
typedef __attribute__((ext_vector_type(8)))  float  v8f;
typedef int v4i __attribute__((vector_size(16)));   // matches builtin's b128 pointee

#define BB 32
#define TT 512
#define DD 512
#define MROWS (BB*TT)   // 16384
#define LDP 72          // padded LDS row stride (bf16 elems): 36 words -> no bank conflicts

#if defined(__has_builtin)
#if __has_builtin(__builtin_amdgcn_global_load_async_to_lds_b128) && \
    __has_builtin(__builtin_amdgcn_s_wait_asynccnt)
#define USE_ASYNC_LDS 1
#endif
#endif
#ifndef USE_ASYNC_LDS
#define USE_ASYNC_LDS 0
#endif

#define ASG __attribute__((address_space(1)))
#define ASL __attribute__((address_space(3)))

__device__ __forceinline__ v16bf make16(v8bf lo, v8bf hi) {
  return __builtin_shufflevector(lo, hi, 0,1,2,3,4,5,6,7,8,9,10,11,12,13,14,15);
}
__device__ __forceinline__ v8f vzero8() {
  v8f z;
#pragma unroll
  for (int i = 0; i < 8; ++i) z[i] = 0.0f;
  return z;
}
__device__ __forceinline__ float gelu_exact(float x) {
  return 0.5f * x * (1.0f + erff(x * 0.70710678118654752f));
}

// ---------------------------------------------------------------------------
// Weight convert + transpose: src f32 (K x N, row-major) -> dst bf16 (N x K)
// ---------------------------------------------------------------------------
__global__ void __launch_bounds__(256) wt_transpose_kernel(
    const float* __restrict__ src, bf16* __restrict__ dst, int K, int N) {
  int idx = blockIdx.x * 256 + threadIdx.x;
  if (idx >= N * K) return;
  int n = idx / K;
  int k = idx - n * K;
  dst[idx] = (bf16)src[(size_t)k * N + n];
}

// ---------------------------------------------------------------------------
// Embedding gathers
// ---------------------------------------------------------------------------
__global__ void __launch_bounds__(256) embed_pitch_kernel(
    const int* __restrict__ p, const int* __restrict__ ts,
    const int* __restrict__ vel, const int* __restrict__ dur,
    const float* __restrict__ pe, const float* __restrict__ te,
    const float* __restrict__ ve, const float* __restrict__ de,
    float* __restrict__ x) {
  int idx = blockIdx.x * 256 + threadIdx.x;  // row*512 + d
  int row = idx >> 9, d = idx & 511;
  x[idx] = pe[(size_t)p[row] * DD + d] + te[(size_t)ts[row] * DD + d] +
           ve[(size_t)vel[row] * DD + d] + de[(size_t)dur[row] * DD + d];
}

__global__ void __launch_bounds__(256) embed_pos_kernel(
    const int* __restrict__ pos, const float* __restrict__ pe_p,
    const float* __restrict__ pe_t, float* __restrict__ x) {
  int idx = blockIdx.x * 256 + threadIdx.x;
  int row = idx >> 9, d = idx & 511;
  int t = row & (TT - 1);
  x[idx] = pe_p[(size_t)pos[row] * DD + d] + pe_t[(size_t)t * DD + d];
}

// ---------------------------------------------------------------------------
// LayerNorm over D=512, f32 in, bf16 out (ldo/col-offset folded into `out`)
// ---------------------------------------------------------------------------
__global__ void __launch_bounds__(128) ln_bf16_kernel(
    const float* __restrict__ x, const float* __restrict__ g,
    const float* __restrict__ b, bf16* __restrict__ out, int ldo) {
  int row = blockIdx.x, tid = threadIdx.x;
  const float* xr = x + (size_t)row * DD;
  float v[4], s = 0.f, s2 = 0.f;
#pragma unroll
  for (int j = 0; j < 4; ++j) {
    v[j] = xr[tid + j * 128];
    s += v[j]; s2 += v[j] * v[j];
  }
  __shared__ float rs[128], rq[128];
  rs[tid] = s; rq[tid] = s2;
  __syncthreads();
  for (int st = 64; st > 0; st >>= 1) {
    if (tid < st) { rs[tid] += rs[tid + st]; rq[tid] += rq[tid + st]; }
    __syncthreads();
  }
  float mean = rs[0] * (1.f / 512.f);
  float var  = rq[0] * (1.f / 512.f) - mean * mean;
  float inv  = rsqrtf(var + 1e-5f);
#pragma unroll
  for (int j = 0; j < 4; ++j) {
    int c = tid + j * 128;
    out[(size_t)row * ldo + c] = (bf16)(((v[j] - mean) * inv) * g[c] + b[c]);
  }
}

// ---------------------------------------------------------------------------
// GELU elementwise (fusion path)
// ---------------------------------------------------------------------------
__global__ void __launch_bounds__(256) gelu_kernel(float* __restrict__ x, int n) {
  int i = blockIdx.x * 256 + threadIdx.x;
  if (i < n) x[i] = gelu_exact(x[i]);
}

// ---------------------------------------------------------------------------
// RoPE: qkv bf16 (M x 1536) -> q,k bf16 (M x 512); v stays in qkv buffer.
// ---------------------------------------------------------------------------
__global__ void __launch_bounds__(256) rope_kernel(
    const bf16* __restrict__ qkv, bf16* __restrict__ qo, bf16* __restrict__ ko) {
  int row = blockIdx.x;
  int pr  = threadIdx.x;           // 0..255
  int t   = row & (TT - 1);
  int d0  = pr * 2;                // 0..510 within 512
  int kidx = d0 & 63;              // even index within head = "k" in reference
  float theta = __powf(10000.f, -((float)kidx) / 64.f);
  float ang = (float)t * theta;
  float sn, cs;
  __sincosf(ang, &sn, &cs);
  const bf16* r = qkv + (size_t)row * 1536;
  float qr = (float)r[d0],        qi = (float)r[d0 + 1];
  float kr = (float)r[512 + d0],  ki = (float)r[512 + d0 + 1];
  bf16* qd = qo + (size_t)row * DD + d0;
  bf16* kd = ko + (size_t)row * DD + d0;
  qd[0] = (bf16)(qr * cs - qi * sn);
  qd[1] = (bf16)(qr * sn + qi * cs);
  kd[0] = (bf16)(kr * cs - ki * sn);
  kd[1] = (bf16)(kr * sn + ki * cs);
}

// ---------------------------------------------------------------------------
// bf16 WMMA GEMM: C(MxN) = A(MxK, lda=K) @ W (weights TRANSPOSED: NxK).
// Block 128x128, BK=64, 8 waves, wave tile 64x32 (4x2 WMMA 16x16x32 accs).
// Double-buffered LDS (one barrier per K-tile); staging via async global->LDS
// DMA when available (ASYNCcnt), else global_load_b128 + ds_store_b128.
// Ragged N handled by clamped row pointers (no EXEC masking in hot loop).
// ---------------------------------------------------------------------------
__global__ void __launch_bounds__(256) gemm_bf16_kernel(
    const bf16* __restrict__ A, const bf16* __restrict__ Wt,
    const float* __restrict__ bias, const float* __restrict__ resid,
    void* __restrict__ outp, int M, int N, int K, int ldo, int act, int out_bf16) {
  __shared__ alignas(16) bf16 As[2][128 * LDP];
  __shared__ alignas(16) bf16 Bs[2][128 * LDP];

  int tid  = threadIdx.x;
  int wave = tid >> 5, lane = tid & 31;
  int hl   = lane >> 4, l16 = lane & 15;
  int wm   = wave >> 2, wn = wave & 3;       // wave tile: M 64 x N 32
  int m0   = blockIdx.y * 128, n0 = blockIdx.x * 128;

  v8f acc[4][2];
#pragma unroll
  for (int a = 0; a < 4; ++a)
#pragma unroll
    for (int c = 0; c < 2; ++c) acc[a][c] = vzero8();

  // per-thread staging coordinates: 4 chunks of 8 bf16 for each of A and B
  int rowA = tid >> 3;            // 0..31 (chunk c adds c*32)
  int kp   = (tid & 7) * 8;       // 0..56
  const bf16* aptr = A + (size_t)(m0 + rowA) * K + kp;
  const bf16* bptr[4];
#pragma unroll
  for (int c = 0; c < 4; ++c) {
    int r = n0 + c * 32 + rowA;
    if (r > N - 1) r = N - 1;     // clamp: garbage only feeds discarded columns
    bptr[c] = Wt + (size_t)r * K + kp;
  }

  auto stage = [&](int buf, int koff) {
#if USE_ASYNC_LDS
#pragma unroll
    for (int c = 0; c < 4; ++c)
      __builtin_amdgcn_global_load_async_to_lds_b128(
          (ASG v4i*)(aptr + koff + (size_t)(c * 32) * K),
          (ASL v4i*)&As[buf][(rowA + c * 32) * LDP + kp], 0, 0);
#pragma unroll
    for (int c = 0; c < 4; ++c)
      __builtin_amdgcn_global_load_async_to_lds_b128(
          (ASG v4i*)(bptr[c] + koff),
          (ASL v4i*)&Bs[buf][(rowA + c * 32) * LDP + kp], 0, 0);
#else
#pragma unroll
    for (int c = 0; c < 4; ++c) {
      v8bf d = *(const v8bf*)(aptr + koff + (size_t)(c * 32) * K);
      *(v8bf*)(&As[buf][(rowA + c * 32) * LDP + kp]) = d;
    }
#pragma unroll
    for (int c = 0; c < 4; ++c) {
      v8bf d = *(const v8bf*)(bptr[c] + koff);
      *(v8bf*)(&Bs[buf][(rowA + c * 32) * LDP + kp]) = d;
    }
#endif
  };

  int nk = K >> 6;
  stage(0, 0);
  for (int kt = 0; kt < nk; ++kt) {
#if USE_ASYNC_LDS
    __builtin_amdgcn_s_wait_asynccnt(0);          // my copies for tile kt landed
#endif
    __syncthreads();                              // everyone's copies visible
    if (kt + 1 < nk) stage((kt + 1) & 1, (kt + 1) * 64);
    const bf16* as = As[kt & 1];
    const bf16* bs = Bs[kt & 1];
#pragma unroll
    for (int s = 0; s < 2; ++s) {                 // two 16x16x32 k-steps
      v16bf af[4], bfv[2];
#pragma unroll
      for (int mt = 0; mt < 4; ++mt) {
        const bf16* ap = as + (wm * 64 + mt * 16 + l16) * LDP + s * 32;
        v8bf lo = *(const v8bf*)(ap + hl * 8);
        v8bf hi = *(const v8bf*)(ap + 16 + hl * 8);
        af[mt] = make16(lo, hi);
      }
#pragma unroll
      for (int nt = 0; nt < 2; ++nt) {
        const bf16* bp = bs + (wn * 32 + nt * 16 + l16) * LDP + s * 32;
        v8bf lo = *(const v8bf*)(bp + hl * 8);
        v8bf hi = *(const v8bf*)(bp + 16 + hl * 8);
        bfv[nt] = make16(lo, hi);
      }
#pragma unroll
      for (int mt = 0; mt < 4; ++mt)
#pragma unroll
        for (int nt = 0; nt < 2; ++nt)
          acc[mt][nt] = __builtin_amdgcn_wmma_f32_16x16x32_bf16(
              false, af[mt], false, bfv[nt], (short)0, acc[mt][nt], false, false);
    }
  }

  // --- epilogue
#pragma unroll
  for (int mt = 0; mt < 4; ++mt)
#pragma unroll
    for (int nt = 0; nt < 2; ++nt)
#pragma unroll
      for (int i = 0; i < 8; ++i) {
        int m = m0 + wm * 64 + mt * 16 + i + hl * 8;
        int n = n0 + wn * 32 + nt * 16 + l16;
        if (n >= N) continue;
        float val = acc[mt][nt][i];
        if (bias) val += bias[n];
        if (act == 1) val = gelu_exact(val);
        else if (act == 2) val = fmaxf(val, 0.0f);
        if (resid) val += resid[(size_t)m * ldo + n];
        if (out_bf16) ((bf16*)outp)[(size_t)m * ldo + n] = (bf16)val;
        else          ((float*)outp)[(size_t)m * ldo + n] = val;
      }
}

// ---------------------------------------------------------------------------
// Flash attention, bf16 inputs, f32 softmax/accum, bf16 output (M x 512).
// grid = (T/64, H, B); block = 128 threads (4 waves), wave owns 16 q-rows.
// q/k/v addressed as row*stride + h*64 + dh. DH = 64.
// ---------------------------------------------------------------------------
__global__ void __launch_bounds__(128) flash_kernel(
    const bf16* __restrict__ q, const bf16* __restrict__ k,
    const bf16* __restrict__ v, bf16* __restrict__ out,
    int sQ, int sK, int sV, int causal) {
  int qt = blockIdx.x, h = blockIdx.y, b = blockIdx.z;
  int wave = threadIdx.x >> 5, lane = threadIdx.x & 31;
  int hl = lane >> 4, l16 = lane & 15;
  int qbase = qt * 64;
  int hoff = h * 64;

  // Q fragments for this wave's 16 rows (K-dim 64 -> two 16x32 A-fragments)
  v16bf qf[2];
  {
    size_t rowQ = (size_t)(b * TT + qbase + wave * 16 + l16);
#pragma unroll
    for (int ks = 0; ks < 2; ++ks) {
      const bf16* p = q + rowQ * sQ + hoff + ks * 32;
      v8bf lo = *(const v8bf*)(p + hl * 8);
      v8bf hi = *(const v8bf*)(p + 16 + hl * 8);
      qf[ks] = make16(lo, hi);
    }
  }

  v8f o[4];
#pragma unroll
  for (int j = 0; j < 4; ++j) o[j] = vzero8();
  float rm[8], rl[8];
#pragma unroll
  for (int i = 0; i < 8; ++i) { rm[i] = -3.0e38f; rl[i] = 0.0f; }

  __shared__ alignas(16) bf16 Vt[64 * LDP];       // Vt[dh][key], padded stride
  __shared__ alignas(16) bf16 Pl[4][16 * LDP];    // per-wave P tile, padded

  int nkt = causal ? (qt + 1) : (TT / 64);
  for (int kt = 0; kt < nkt; ++kt) {
    int kb = kt * 64;
    // cooperative transposed V load: thread -> (row r, half of dh)
    {
      int r = threadIdx.x >> 1;
      int part = (threadIdx.x & 1) * 32;
      const bf16* vp = v + (size_t)(b * TT + kb + r) * sV + hoff + part;
#pragma unroll
      for (int c4 = 0; c4 < 4; ++c4) {
        v8bf d = *(const v8bf*)(vp + c4 * 8);
#pragma unroll
        for (int e = 0; e < 8; ++e) Vt[(part + c4 * 8 + e) * LDP + r] = d[e];
      }
    }
    __syncthreads();

    // S = Q @ K^T (per wave: 16 x 64)
    v8f s[4];
#pragma unroll
    for (int j = 0; j < 4; ++j) {
      v8f a = vzero8();
      size_t rowK = (size_t)(b * TT + kb + j * 16 + l16);
#pragma unroll
      for (int ks = 0; ks < 2; ++ks) {
        const bf16* p = k + rowK * sK + hoff + ks * 32;
        v8bf lo = *(const v8bf*)(p + hl * 8);
        v8bf hi = *(const v8bf*)(p + 16 + hl * 8);
        a = __builtin_amdgcn_wmma_f32_16x16x32_bf16(
            false, qf[ks], false, make16(lo, hi), (short)0, a, false, false);
      }
      s[j] = a;
    }
#pragma unroll
    for (int j = 0; j < 4; ++j)
#pragma unroll
      for (int i = 0; i < 8; ++i) s[j][i] *= 0.125f;  // DH^-0.5

    if (causal && kt == qt) {
#pragma unroll
      for (int j = 0; j < 4; ++j)
#pragma unroll
        for (int i = 0; i < 8; ++i) {
          int mrow = qbase + wave * 16 + i + hl * 8;
          int ncol = kb + j * 16 + l16;
          if (ncol > mrow) s[j][i] = -3.0e38f;
        }
    }

    // online softmax; each row lives in one 16-lane half -> xor-reduce 1,2,4,8
    float pbuf[4][8];
#pragma unroll
    for (int i = 0; i < 8; ++i) {
      float tm = s[0][i];
#pragma unroll
      for (int j = 1; j < 4; ++j) tm = fmaxf(tm, s[j][i]);
#pragma unroll
      for (int mm = 1; mm < 16; mm <<= 1) tm = fmaxf(tm, __shfl_xor(tm, mm, 32));
      float nm = fmaxf(rm[i], tm);
      float fi = __expf(rm[i] - nm);
      float tsum = 0.0f;
#pragma unroll
      for (int j = 0; j < 4; ++j) {
        float e = __expf(s[j][i] - nm);
        pbuf[j][i] = e; tsum += e;
      }
#pragma unroll
      for (int mm = 1; mm < 16; mm <<= 1) tsum += __shfl_xor(tsum, mm, 32);
      rl[i] = rl[i] * fi + tsum;
      rm[i] = nm;
#pragma unroll
      for (int j = 0; j < 4; ++j) o[j][i] *= fi;
    }

    // P (C-layout) -> LDS -> A-fragment layout
#pragma unroll
    for (int j = 0; j < 4; ++j)
#pragma unroll
      for (int i = 0; i < 8; ++i)
        Pl[wave][(i + hl * 8) * LDP + j * 16 + l16] = (bf16)pbuf[j][i];
    __syncthreads();

    // O += P @ V
#pragma unroll
    for (int ks = 0; ks < 2; ++ks) {
      const bf16* pp = &Pl[wave][l16 * LDP + ks * 32];
      v8bf plo = *(const v8bf*)(pp + hl * 8);
      v8bf phi = *(const v8bf*)(pp + 16 + hl * 8);
      v16bf pf = make16(plo, phi);
#pragma unroll
      for (int j = 0; j < 4; ++j) {
        const bf16* vp2 = &Vt[(j * 16 + l16) * LDP + ks * 32];
        v8bf vlo = *(const v8bf*)(vp2 + hl * 8);
        v8bf vhi = *(const v8bf*)(vp2 + 16 + hl * 8);
        o[j] = __builtin_amdgcn_wmma_f32_16x16x32_bf16(
            false, pf, false, make16(vlo, vhi), (short)0, o[j], false, false);
      }
    }
    __syncthreads();
  }

  // normalize + store bf16 into (M x 512) at head column block
#pragma unroll
  for (int i = 0; i < 8; ++i) rl[i] = 1.0f / rl[i];
#pragma unroll
  for (int j = 0; j < 4; ++j)
#pragma unroll
    for (int i = 0; i < 8; ++i) {
      int t = qbase + wave * 16 + i + hl * 8;
      float val = o[j][i] * rl[i];
      out[(size_t)(b * TT + t) * DD + hoff + j * 16 + l16] = (bf16)val;
    }
}

// ---------------------------------------------------------------------------
// Host-side orchestration.
// Input order (setup_inputs insertion order):
//  0..4  pitches, time_shifts, velocities, durations, positions (int, 16384)
//  5..8  pitch_embed, ts_embed, vel_embed, dur_embed
//  9..48 pitch_layers[4] x {ln1_g,ln1_b,qkv_w,proj_w,ln2_g,ln2_b,ff1_w,ff1_b,ff2_w,ff2_b}
//  49,50 pitch_norm_g/b ; 51 pos_embed_t ; 52 pos_embed_p
//  53..76 pos_layers[2] x {ln1_g,ln1_b,qkv_w,qkv_b,out_w,out_b,ln2_g,ln2_b,ff1_w,ff1_b,ff2_w,ff2_b}
//  77,78 pos_norm_g/b ; 79,80 fusion_w/b ; 81,82 fusion_ln_g/b
//  83..90 head_w/b, dur_w/b, tonic_w/b, pitch_head_w/b
// ---------------------------------------------------------------------------
extern "C" void kernel_launch(void* const* d_in, const int* in_sizes, int n_in,
                              void* d_out, int out_size, void* d_ws, size_t ws_size,
                              hipStream_t stream) {
  (void)in_sizes; (void)n_in; (void)out_size; (void)ws_size;
  const int M = MROWS;

  auto F = [&](int i) { return (const float*)d_in[i]; };
  const int* pitches = (const int*)d_in[0];
  const int* tshifts = (const int*)d_in[1];
  const int* vels    = (const int*)d_in[2];
  const int* durs    = (const int*)d_in[3];
  const int* poss    = (const int*)d_in[4];

  char* ws = (char*)d_ws;
  size_t off = 0;
  auto alloc = [&](size_t bytes) -> void* {
    off = (off + 255) & ~(size_t)255;
    void* p = ws + off;
    off += bytes;
    return p;
  };
  // convert + transpose weight: src f32 (K x N) -> bf16 (N x K)
  auto tobfT = [&](const float* src, int K, int N) -> bf16* {
    size_t n = (size_t)K * N;
    bf16* dst = (bf16*)alloc(n * sizeof(bf16));
    int blocks = (int)((n + 255) / 256);
    wt_transpose_kernel<<<blocks, 256, 0, stream>>>(src, dst, K, N);
    return dst;
  };

  float* xp     = (float*)alloc((size_t)M * 512 * 4);
  float* xq     = (float*)alloc((size_t)M * 512 * 4);
  bf16*  hb     = (bf16*)alloc((size_t)M * 512 * 2);
  bf16*  qkvb   = (bf16*)alloc((size_t)M * 1536 * 2);
  bf16*  qb     = (bf16*)alloc((size_t)M * 512 * 2);
  bf16*  kb     = (bf16*)alloc((size_t)M * 512 * 2);
  bf16*  aob    = (bf16*)alloc((size_t)M * 512 * 2);
  bf16*  ffb    = (bf16*)alloc((size_t)M * 2048 * 2);
  bf16*  catb   = (bf16*)alloc((size_t)M * 1024 * 2);
  float* fusedf = (float*)alloc((size_t)M * 512 * 4);
  bf16*  fusedb = (bf16*)alloc((size_t)M * 512 * 2);

  auto ln = [&](const float* x, const float* g, const float* bta, bf16* out, int ldo) {
    ln_bf16_kernel<<<M, 128, 0, stream>>>(x, g, bta, out, ldo);
  };
  auto gemm = [&](const bf16* A, const bf16* Wt, int N, int K, const float* bias,
                  int act, const float* resid, void* out, int obf, int ldo) {
    dim3 g((N + 127) / 128, M / 128);
    gemm_bf16_kernel<<<g, 256, 0, stream>>>(A, Wt, bias, resid, out, M, N, K, ldo, act, obf);
  };

  // embeddings
  embed_pitch_kernel<<<(M * 512) / 256, 256, 0, stream>>>(
      pitches, tshifts, vels, durs, F(5), F(6), F(7), F(8), xp);
  embed_pos_kernel<<<(M * 512) / 256, 256, 0, stream>>>(poss, F(52), F(51), xq);

  // ---- pitch encoder (4 layers, RoPE, full attention, GELU FFN)
  for (int l = 0; l < 4; ++l) {
    int b0 = 9 + l * 10;
    bf16* qkvw  = tobfT(F(b0 + 2), 512, 1536);
    bf16* projw = tobfT(F(b0 + 3), 512, 512);
    bf16* ff1w  = tobfT(F(b0 + 6), 512, 2048);
    bf16* ff2w  = tobfT(F(b0 + 8), 2048, 512);

    ln(xp, F(b0 + 0), F(b0 + 1), hb, 512);
    gemm(hb, qkvw, 1536, 512, nullptr, 0, nullptr, qkvb, 1, 1536);
    rope_kernel<<<M, 256, 0, stream>>>(qkvb, qb, kb);
    {
      dim3 g(TT / 64, 8, BB);
      flash_kernel<<<g, 128, 0, stream>>>(qb, kb, qkvb + 1024, aob, 512, 512, 1536, 0);
    }
    gemm(aob, projw, 512, 512, nullptr, 0, xp, xp, 0, 512);
    ln(xp, F(b0 + 4), F(b0 + 5), hb, 512);
    gemm(hb, ff1w, 2048, 512, F(b0 + 7), /*gelu*/1, nullptr, ffb, 1, 2048);
    gemm(ffb, ff2w, 512, 2048, F(b0 + 9), 0, xp, xp, 0, 512);
  }

  // ---- pos encoder (2 layers, causal, ReLU FFN)
  for (int l = 0; l < 2; ++l) {
    int b0 = 53 + l * 12;
    bf16* qkvw = tobfT(F(b0 + 2), 512, 1536);
    bf16* outw = tobfT(F(b0 + 4), 512, 512);
    bf16* ff1w = tobfT(F(b0 + 8), 512, 1024);
    bf16* ff2w = tobfT(F(b0 + 10), 1024, 512);

    ln(xq, F(b0 + 0), F(b0 + 1), hb, 512);
    gemm(hb, qkvw, 1536, 512, F(b0 + 3), 0, nullptr, qkvb, 1, 1536);
    {
      dim3 g(TT / 64, 8, BB);
      flash_kernel<<<g, 128, 0, stream>>>(qkvb, qkvb + 512, qkvb + 1024, aob,
                                          1536, 1536, 1536, 1);
    }
    gemm(aob, outw, 512, 512, F(b0 + 5), 0, xq, xq, 0, 512);
    ln(xq, F(b0 + 6), F(b0 + 7), hb, 512);
    gemm(hb, ff1w, 1024, 512, F(b0 + 9), /*relu*/2, nullptr, ffb, 1, 1024);
    gemm(ffb, ff2w, 512, 1024, F(b0 + 11), 0, xq, xq, 0, 512);
  }

  // ---- fusion + heads
  ln(xp, F(49), F(50), catb, 1024);          // pitch_norm -> concat[:, :512]
  ln(xq, F(77), F(78), catb + 512, 1024);    // pos_norm   -> concat[:, 512:]
  bf16* fw = tobfT(F(79), 1024, 512);
  gemm(catb, fw, 512, 1024, F(80), 0, nullptr, fusedf, 0, 512);
  gelu_kernel<<<(M * 512) / 256, 256, 0, stream>>>(fusedf, M * 512);
  ln(fusedf, F(81), F(82), fusedb, 512);

  float* oh = (float*)d_out;
  bf16* hw = tobfT(F(83), 512, 126);
  gemm(fusedb, hw, 126, 512, F(84), 0, nullptr, oh, 0, 126);
  bf16* dw = tobfT(F(85), 512, 17);
  gemm(fusedb, dw, 17, 512, F(86), 0, nullptr, oh + (size_t)M * 126, 0, 17);
  bf16* tw = tobfT(F(87), 512, 12);
  gemm(fusedb, tw, 12, 512, F(88), 0, nullptr, oh + (size_t)M * (126 + 17), 0, 12);
  bf16* pw = tobfT(F(89), 512, 128);
  gemm(fusedb, pw, 128, 512, F(90), 0, nullptr, oh + (size_t)M * (126 + 17 + 12), 0, 128);
}
